// LinearTripleRCell_62285615727033
// MI455X (gfx1250) — matrix operand order repaired and checked by reading, not compiled, not run on hardware
//
#include <hip/hip_runtime.h>
#include <hip/hip_bf16.h>
#include <math.h>

// ---------------------------------------------------------------------------
// Types
// ---------------------------------------------------------------------------
typedef __attribute__((ext_vector_type(16))) __bf16 v16bf;
typedef __attribute__((ext_vector_type(8)))  __bf16 v8bf;
typedef __attribute__((ext_vector_type(4)))  __bf16 v4bf;
typedef __attribute__((ext_vector_type(8)))  float  v8f;
typedef __attribute__((ext_vector_type(4)))  unsigned int u32x4;
typedef __attribute__((ext_vector_type(8)))  int    i32x8;
typedef __attribute__((ext_vector_type(4)))  int    i32x4;

#define DIM   1024
#define TT    1024
#define BB    32
#define ROWS  (TT * BB)          // 32768
#define H_ELEMS ((TT + 1) * BB * DIM)

// ---------------------------------------------------------------------------
// TDM: async tensor DMA of a [tile_rows x tile_k] bf16 tile (row stride
// row_stride_elems in global) into LDS.  pad=1 inserts 16B of LDS padding per
// 64B row (tile_k==32), giving an 80B LDS row pitch (16B aligned, bank-spread).
// Descriptor bitfields per CDNA5 ISA ch.8 (groups 0/1; 2D tile, groups 2/3 = 0).
// This toolchain exposes the 6-arg builtin:
//   (u32x4 g0, i32x8 g1, i32x4 g2, i32x4 g3, i32x8 g4, i32 cpol)
// ---------------------------------------------------------------------------
__device__ __forceinline__ void tdm_load_tile(unsigned lds_addr, const void* gptr,
                                              int tile_k, int tile_rows,
                                              unsigned long long row_stride_elems,
                                              int pad) {
  unsigned long long ga = (unsigned long long)gptr;
  u32x4 g0;
  g0[0] = 1u;                                             // count=1, user desc
  g0[1] = lds_addr;                                       // lds_addr [63:32]
  g0[2] = (unsigned)ga;                                   // global_addr[31:0]
  g0[3] = ((unsigned)(ga >> 32) & 0x01FFFFFFu)            // global_addr[56:32]
        | 0x80000000u;                                    // type=2 ("image")
  unsigned long long td0 = 1ull << 20, td1 = 1ull << 20;  // generous dims (no OOB)
  unsigned long long s0  = row_stride_elems;
  i32x8 g1;
  g1[0] = (int)((1u << 16) |                              // data_size = 2B
                (pad ? ((1u << 20) | (3u << 22) | (3u << 25)) : 0u)); // pad 16B/64B
  g1[1] = (int)((td0 & 0xFFFFu) << 16);                   // tensor_dim0 lo
  g1[2] = (int)(((td0 >> 16) & 0xFFFFu) | ((td1 & 0xFFFFu) << 16));
  g1[3] = (int)(((td1 >> 16) & 0xFFFFu) | (((unsigned)tile_k & 0xFFFFu) << 16));
  g1[4] = (int)((unsigned)tile_rows & 0xFFFFu);           // tile_dim1 (tile_dim2=0)
  g1[5] = (int)(s0 & 0xFFFFFFFFu);                        // dim0 stride lo32
  g1[6] = (int)((s0 >> 32) & 0xFFFFu);                    // dim0 stride hi16
  g1[7] = 0;
  i32x4 g2 = {0, 0, 0, 0}, g3 = {0, 0, 0, 0};
  i32x8 g4 = {0, 0, 0, 0, 0, 0, 0, 0};
  __builtin_amdgcn_tensor_load_to_lds(g0, g1, g2, g3, g4, 0);
}

// ---------------------------------------------------------------------------
// WMMA fragment loads (bf16, 16x16x32). Per CDNA5 ISA 7.12.2:
//  A (16x32 MxK): lanes 0-15 -> K[0..7],K[16..23]; lanes 16-31 -> K[8..15],K[24..31]
//  B (32x16 KxN): lanes 0-15 -> K[0..7],K[16..23]; lanes 16-31 -> K[8..15],K[24..31]
// Both map to two 16B contiguous-K loads per lane for row-major [row][K] tiles.
// `base` already points at (row0, k0) of the tile; rowPitch in bytes.
// ---------------------------------------------------------------------------
__device__ __forceinline__ v16bf load_a_frag(const unsigned char* base, int rowPitch, int lane) {
  const unsigned char* p = base + (lane & 15) * rowPitch + (lane >> 4) * 16;
  v16bf f;
  ((v8bf*)&f)[0] = *(const v8bf*)(p);
  ((v8bf*)&f)[1] = *(const v8bf*)(p + 32);
  return f;
}
__device__ __forceinline__ v16bf load_b_frag(const unsigned char* base, int rowPitch, int lane) {
  const unsigned char* p = base + (lane & 15) * rowPitch + (lane >> 4) * 16;
  v16bf f;
  ((v8bf*)&f)[0] = *(const v8bf*)(p);
  ((v8bf*)&f)[1] = *(const v8bf*)(p + 32);
  return f;
}
__device__ __forceinline__ v8f wmma_bf16(v16bf a, v16bf b, v8f c) {
  return __builtin_amdgcn_wmma_f32_16x16x32_bf16(false, a, false, b, (short)0, c,
                                                 false, false);
}

// ---------------------------------------------------------------------------
// Software grid barrier (sense via generation counter in ws).
// ---------------------------------------------------------------------------
__device__ __forceinline__ void grid_barrier(unsigned* bar, unsigned nblk) {
  __threadfence();
  __syncthreads();
  if (threadIdx.x == 0) {
    unsigned gen = __atomic_load_n(bar + 1, __ATOMIC_ACQUIRE);
    if (atomicAdd(bar, 1u) == nblk - 1u) {
      __atomic_store_n(bar, 0u, __ATOMIC_RELAXED);
      __atomic_fetch_add(bar + 1, 1u, __ATOMIC_ACQ_REL);
    } else {
      while (__atomic_load_n(bar + 1, __ATOMIC_ACQUIRE) == gen)
        __builtin_amdgcn_s_sleep(2);
    }
  }
  __syncthreads();
}

// ---------------------------------------------------------------------------
// Precision conversion kernels (fp32 -> bf16 working copies)
// ---------------------------------------------------------------------------
__global__ __launch_bounds__(256) void convert_weights_kernel(
    const float* Rx, const float* Wd, const float* Rh, const float* Rd,
    const float* Wo, const float* h0, __bf16* wbf, float* h_out, __bf16* h_ping) {
  int i = blockIdx.x * blockDim.x + threadIdx.x;
  const int n = 5 * DIM * DIM;
  if (i < n) {
    int sel = i >> 20, off = i & (DIM * DIM - 1);
    const float* src = (sel == 0) ? Rx : (sel == 1) ? Wd : (sel == 2) ? Rh
                     : (sel == 3) ? Rd : Wo;
    wbf[i] = (__bf16)src[off];
  }
  if (i < BB * DIM) {           // h[0] = h0 (fp32 out) + bf16 ping buffer
    float v = h0[i];
    h_out[i] = v;
    h_ping[i] = (__bf16)v;
  }
}

__global__ __launch_bounds__(256) void convert_x_kernel(const float* x, __bf16* xbf) {
  int i = (blockIdx.x * blockDim.x + threadIdx.x) * 4;
  if (i < ROWS * DIM) {
    float4 v = *(const float4*)(x + i);
    v4bf o;
    o[0] = (__bf16)v.x; o[1] = (__bf16)v.y; o[2] = (__bf16)v.z; o[3] = (__bf16)v.w;
    *(v4bf*)(xbf + i) = o;
  }
}

// ---------------------------------------------------------------------------
// Big NT GEMM: C[r,e] = sum_k A[r,k]*B[e,k] (+bias[e]).  A:[32768x1024] bf16,
// B:[1024x1024] bf16, C fp32.  128x128 macro tile, 8 waves (2Mx4N, 64x32 each),
// K staged 32 at a time via double-buffered TDM tiles (80B LDS row pitch).
// blockIdx.z selects (B0,C0,bias0) vs (B1,C1,bias1).
// ---------------------------------------------------------------------------
__global__ __launch_bounds__(256) void gemm_nt_bf16_kernel(
    const __bf16* A, const __bf16* B0, const __bf16* B1,
    float* C0, float* C1, const float* bias0, const float* bias1) {
  const __bf16* Bm = blockIdx.z ? B1 : B0;
  float* C = blockIdx.z ? C1 : C0;
  const float* bias = blockIdx.z ? bias1 : bias0;

  __shared__ unsigned char ldsA[2][128 * 80];
  __shared__ unsigned char ldsB[2][128 * 80];

  const int tid = threadIdx.x, lane = tid & 31, wid = tid >> 5;
  const int m0 = blockIdx.x * 128, n0 = blockIdx.y * 128;
  const int mg = wid >> 2, ng = wid & 3;    // wave tile: rows [64*mg,+64), cols [32*ng,+32)

  const v8f vz = {0.f, 0.f, 0.f, 0.f, 0.f, 0.f, 0.f, 0.f};
  v8f acc[4][2];
#pragma unroll
  for (int ms = 0; ms < 4; ++ms)
#pragma unroll
    for (int ns = 0; ns < 2; ++ns) acc[ms][ns] = vz;

  // each wave DMAs its own 16-row slice of the A and B k-chunk tiles
  auto issue = [&](int kt, int buf) {
    tdm_load_tile((unsigned)(uintptr_t)&ldsA[buf][wid * 16 * 80],
                  A + (size_t)(m0 + wid * 16) * DIM + kt * 32, 32, 16, DIM, 1);
    tdm_load_tile((unsigned)(uintptr_t)&ldsB[buf][wid * 16 * 80],
                  Bm + (size_t)(n0 + wid * 16) * DIM + kt * 32, 32, 16, DIM, 1);
  };

  issue(0, 0);
  for (int kt = 0; kt < 32; ++kt) {
    const int cur = kt & 1;
    if (kt + 1 < 32) {
      issue(kt + 1, cur ^ 1);
      __builtin_amdgcn_s_wait_tensorcnt(2);   // oldest pair (current buf) done
    } else {
      __builtin_amdgcn_s_wait_tensorcnt(0);
    }
    __syncthreads();                          // all waves' slices resident

    v16bf af[4];
#pragma unroll
    for (int ms = 0; ms < 4; ++ms)
      af[ms] = load_a_frag(&ldsA[cur][(mg * 64 + ms * 16) * 80], 80, lane);
#pragma unroll
    for (int ns = 0; ns < 2; ++ns) {
      v16bf bf = load_b_frag(&ldsB[cur][(ng * 32 + ns * 16) * 80], 80, lane);
#pragma unroll
      for (int ms = 0; ms < 4; ++ms)
        acc[ms][ns] = wmma_bf16(af[ms], bf, acc[ms][ns]);
    }
    __syncthreads();                          // done reading before next DMA overwrite
  }

  // epilogue: C layout — lane(0-15): M=j, lane(16-31): M=8+j; N = lane&15
#pragma unroll
  for (int ns = 0; ns < 2; ++ns) {
    const int col = n0 + ng * 32 + ns * 16 + (lane & 15);
    const float bv = bias ? bias[col] : 0.f;
#pragma unroll
    for (int ms = 0; ms < 4; ++ms) {
      const int rbase = m0 + mg * 64 + ms * 16 + ((lane >> 4) ? 8 : 0);
#pragma unroll
      for (int j = 0; j < 8; ++j)
        C[(size_t)(rbase + j) * DIM + col] = acc[ms][ns][j] + bv;
    }
  }
}

// ---------------------------------------------------------------------------
// Persistent recurrence kernel.  128 blocks: block g -> matrix (g>>6: R_h /
// R_delta), e-slice 16*(g&63).  Weight slice [16 x 1024] bf16 held in LDS
// (2064B pitch) for all 1024 steps.  Per step: y = h_t @ Wslice^T via WMMA,
// K split 4-ways across waves + LDS reduction; grid barrier; blocks of mat 0
// apply the gated update elementwise; grid barrier.
// ---------------------------------------------------------------------------
__global__ __launch_bounds__(256) void recurrent_kernel(
    const float* xrx, const float* xwd, const __bf16* w_rh, const __bf16* w_rd,
    float* h_out, __bf16* h_ping, __bf16* hs_bf, float* ybuf, unsigned* bar) {
  const int g = blockIdx.x;
  const int mat = g >> 6;
  const int e0 = (g & 63) * 16;
  const int tid = threadIdx.x, lane = tid & 31, wid = tid >> 5;

  __shared__ unsigned char ldsW[16 * 2064];
  __shared__ float red[6 * 256];

  const __bf16* W = mat ? w_rd : w_rh;
  for (int i = tid; i < 16 * 128; i += 256) {          // 128 x 16B chunks per row
    int r = i >> 7, c = i & 127;
    *(uint4*)(ldsW + r * 2064 + c * 16) =
        *(const uint4*)((const unsigned char*)(W + (size_t)(e0 + r) * DIM) + c * 16);
  }
  __syncthreads();

  const int tm = wid & 1;                              // M tile (rows 0-15 / 16-31)
  const int ks = wid >> 1;                             // K quarter (256 each)
  const unsigned char* hbase = (const unsigned char*)h_ping;

  for (int t = 0; t < TT; ++t) {
    const v8f vz = {0.f, 0.f, 0.f, 0.f, 0.f, 0.f, 0.f, 0.f};
    v8f acc = vz;
#pragma unroll
    for (int kk = 0; kk < 8; ++kk) {
      const int kbyte = (ks * 256 + kk * 32) * 2;
      v16bf a = load_a_frag(hbase + tm * 16 * 2048 + kbyte, 2048, lane);
      v16bf b = load_b_frag(ldsW + kbyte, 2064, lane);
      acc = wmma_bf16(a, b, acc);
    }
    if (ks > 0) {
      float* dst = red + (wid - 2) * 256;
#pragma unroll
      for (int j = 0; j < 8; ++j) dst[j * 32 + lane] = acc[j];
    }
    __syncthreads();
    if (ks == 0) {
#pragma unroll
      for (int w = 0; w < 3; ++w) {
        const float* src = red + (tm + w * 2) * 256;
#pragma unroll
        for (int j = 0; j < 8; ++j) acc[j] += src[j * 32 + lane];
      }
#pragma unroll
      for (int j = 0; j < 8; ++j) {
        int row = tm * 16 + ((lane >> 4) ? 8 : 0) + j;
        ybuf[(size_t)(mat * 32 + row) * DIM + e0 + (lane & 15)] = acc[j];
      }
    }
    grid_barrier(bar, 128);

    if (mat == 0) {                                    // gated update for e-slice
      const float* hprev = h_out + (size_t)t * (BB * DIM);
      float* hnext = h_out + (size_t)(t + 1) * (BB * DIM);
      for (int i = tid; i < BB * 16; i += 256) {
        int bb = i >> 4, e = e0 + (i & 15);
        int idx = bb * DIM + e;
        float yh = ybuf[idx];
        float yd = ybuf[BB * DIM + idx];
        float xr = xrx[(size_t)t * (BB * DIM) + idx];
        float xw = xwd[(size_t)t * (BB * DIM) + idx];
        float hp = hprev[idx];
        float cand = tanhf(xr + yh);
        float dl = 1.f / (1.f + __expf(-(xw + yd)));
        float hn = (1.f - dl) * hp + dl * cand;
        hnext[idx] = hn;
        __bf16 hb = (__bf16)hn;
        h_ping[idx] = hb;
        hs_bf[(size_t)(t * BB + bb) * DIM + e] = hb;
      }
    }
    grid_barrier(bar, 128);
  }
}

// ---------------------------------------------------------------------------
// Output epilogue: out = softmax_group32(h[t+1]) * silu(G).  One wave32 per
// (t, b, group) — GROUP_SIZE==32 maps exactly onto a CDNA5 wave.
// ---------------------------------------------------------------------------
__global__ __launch_bounds__(256) void out_elementwise_kernel(
    const float* h, const float* G, float* out) {
  const int lane = threadIdx.x & 31;
  const int wave = (blockIdx.x * blockDim.x + threadIdx.x) >> 5;
  const int nwaves = (gridDim.x * blockDim.x) >> 5;
  for (int task = wave; task < TT * BB * 32; task += nwaves) {
    int t = task >> 10;
    int b = (task >> 5) & 31;
    int grp = task & 31;
    int idx = b * DIM + grp * 32 + lane;
    float hv = h[(size_t)(t + 1) * (BB * DIM) + idx];
    float m = hv;
#pragma unroll
    for (int off = 16; off; off >>= 1) m = fmaxf(m, __shfl_xor(m, off, 32));
    float ex = __expf(hv - m);
    float s = ex;
#pragma unroll
    for (int off = 16; off; off >>= 1) s += __shfl_xor(s, off, 32);
    float gv = G[(size_t)t * (BB * DIM) + idx];
    float silu = gv / (1.f + __expf(-gv));
    out[(size_t)t * (BB * DIM) + idx] = (ex / s) * silu;
  }
}

// ---------------------------------------------------------------------------
// Launch
// ---------------------------------------------------------------------------
extern "C" void kernel_launch(void* const* d_in, const int* in_sizes, int n_in,
                              void* d_out, int out_size, void* d_ws, size_t ws_size,
                              hipStream_t stream) {
  const float* x   = (const float*)d_in[0];
  const float* h0  = (const float*)d_in[1];
  const float* R_h = (const float*)d_in[2];
  const float* R_x = (const float*)d_in[3];
  const float* R_d = (const float*)d_in[4];
  const float* W_d = (const float*)d_in[5];
  const float* W_o = (const float*)d_in[6];
  const float* b   = (const float*)d_in[7];
  const float* b_d = (const float*)d_in[8];
  float* out = (float*)d_out;
  unsigned char* ws = (unsigned char*)d_ws;

  // workspace layout (bytes)
  const size_t OFF_BAR   = 0;                              // 256
  const size_t OFF_YBUF  = 256;                            // 2*32*1024*4 = 256KB
  const size_t OFF_WBF   = OFF_YBUF + 2ull * BB * DIM * 4; // 5*1M bf16 = 10MB
  const size_t OFF_HPING = OFF_WBF + 5ull * DIM * DIM * 2; // 64KB
  const size_t OFF_XBF   = OFF_HPING + (size_t)BB * DIM * 2; // 64MB (reused as hs_bf)
  const size_t OFF_XRX   = OFF_XBF + (size_t)ROWS * DIM * 2; // 128MB (reused as G)
  const size_t OFF_XWD   = OFF_XRX + (size_t)ROWS * DIM * 4; // 128MB

  unsigned* bar   = (unsigned*)(ws + OFF_BAR);
  float*   ybuf   = (float*)(ws + OFF_YBUF);
  __bf16*  wbf    = (__bf16*)(ws + OFF_WBF);
  __bf16*  h_ping = (__bf16*)(ws + OFF_HPING);
  __bf16*  x_bf   = (__bf16*)(ws + OFF_XBF);     // phase 1 input; dead after
  __bf16*  hs_bf  = (__bf16*)(ws + OFF_XBF);     // phase 2 output (reuse)
  float*   xrx    = (float*)(ws + OFF_XRX);      // phase 1 out; dead after scan
  float*   Gbuf   = (float*)(ws + OFF_XRX);      // phase 3 GEMM out (reuse)
  float*   xwd    = (float*)(ws + OFF_XWD);

  __bf16* wRx = wbf;
  __bf16* wWd = wbf + 1ull * DIM * DIM;
  __bf16* wRh = wbf + 2ull * DIM * DIM;
  __bf16* wRd = wbf + 3ull * DIM * DIM;
  __bf16* wWo = wbf + 4ull * DIM * DIM;

  (void)hipMemsetAsync(ws + OFF_BAR, 0, 256, stream);      // grid-barrier state

  convert_weights_kernel<<<(5 * DIM * DIM + 255) / 256, 256, 0, stream>>>(
      R_x, W_d, R_h, R_d, W_o, h0, wbf, out /* h[0] */, h_ping);
  convert_x_kernel<<<(ROWS * DIM / 4 + 255) / 256, 256, 0, stream>>>(x, x_bf);

  // phase 1: xRx = x@R_x^T + b ; xWd = x@W_delta^T + b_delta
  gemm_nt_bf16_kernel<<<dim3(ROWS / 128, DIM / 128, 2), 256, 0, stream>>>(
      x_bf, wRx, wWd, xrx, xwd, b, b_d);

  // phase 2: 1024-step recurrence (persistent, grid-synced)
  recurrent_kernel<<<128, 256, 0, stream>>>(xrx, xwd, wRh, wRd, out, h_ping,
                                            hs_bf, ybuf, bar);

  // phase 3: G = h[1..T]@W_out^T ; out = softmax_group(h) * silu(G)
  gemm_nt_bf16_kernel<<<dim3(ROWS / 128, DIM / 128, 1), 256, 0, stream>>>(
      hs_bf, wWo, wWo, Gbuf, Gbuf, nullptr, nullptr);
  out_elementwise_kernel<<<2048, 256, 0, stream>>>(out, Gbuf, out + H_ELEMS);
}
